// QuantV2XFusionBlock_15899968930303
// MI455X (gfx1250) — compile-verified, matrix operand order checked by reading, not run on hardware
//
#include <hip/hip_runtime.h>
#include <cstddef>

// ---------------- problem dims ----------------
#define HW        17600       // 100*176 BEV positions
#define NTILE_HW  1100        // HW / 16
#define L_AG      5
#define C_CH      256
#define T_TYPES   3
#define HEADS     8

typedef __attribute__((ext_vector_type(16))) __bf16 v16bf;
typedef __attribute__((ext_vector_type(8)))  float  v8f;
typedef __attribute__((ext_vector_type(4)))  int    v4i;

union Frag16 { v16bf v; unsigned int d[8]; unsigned short u[16]; };

__device__ inline unsigned short f2bf(float f) {
  unsigned int u = __float_as_uint(f);
  u += 0x7FFFu + ((u >> 16) & 1u);          // round-to-nearest-even
  return (unsigned short)(u >> 16);
}
__device__ inline float bf2f(unsigned short h) {
  return __uint_as_float(((unsigned int)h) << 16);
}

// ---------------- CDNA5 async global->LDS copy (ASYNCcnt path) ----------------
#if defined(__gfx1250__) && __has_builtin(__builtin_amdgcn_global_load_async_to_lds_b128) && \
    __has_builtin(__builtin_amdgcn_s_wait_asynccnt)
#define HAVE_ASYNC 1
__device__ inline void async_cp16(const unsigned short* g, unsigned short* l) {
  __builtin_amdgcn_global_load_async_to_lds_b128(
      (__attribute__((address_space(1))) v4i*)g,
      (__attribute__((address_space(3))) v4i*)l, 0, 0);
}
__device__ inline void async_wait0() { __builtin_amdgcn_s_wait_asynccnt(0); }
#else
#define HAVE_ASYNC 0
#endif

// Build a 16x32 bf16 WMMA fragment (A: M rows / B: N rows of transposed weights).
// Per ISA 7.12.2: lane = (row & 15) + 16*(K-half); dword slot s holds K = kb + 2s (+8 for s>=4).
// rowp points at the start of this lane's row (K-contiguous bf16), kbase = K-tile offset.
__device__ inline v16bf load_frag(const unsigned short* rowp, int kbase, int kb) {
  Frag16 f;
  const unsigned int* p = (const unsigned int*)rowp;
#pragma unroll
  for (int s = 0; s < 8; ++s) {
    int K = kbase + kb + 2 * s + ((s >= 4) ? 8 : 0);
    f.d[s] = p[K >> 1];
  }
  return f.v;
}

__device__ inline v8f wmma_bf16(v16bf a, v16bf b, v8f c) {
  return __builtin_amdgcn_wmma_f32_16x16x32_bf16(false, a, false, b, (short)0, c,
                                                 false, false);
}

// ---------------- stage 0: weight prep (f32 -> bf16, transposed) ----------------
// wqkvT[t][n(768)][k(256)]  (n: 0..255=Q, 256..511=K, 512..767=V columns)
// waT  [t][n(256)][k(256)]
// raT/rmT[et*8+m][q(32)][p(32)]   (transposed 32x32 relation matrices)
__global__ __launch_bounds__(256) void prep_kernel(
    const float* __restrict__ Wq, const float* __restrict__ Wk,
    const float* __restrict__ Wv, const float* __restrict__ Wa,
    const float* __restrict__ ra, const float* __restrict__ rm,
    unsigned short* __restrict__ wqkvT, unsigned short* __restrict__ waT,
    unsigned short* __restrict__ raT, unsigned short* __restrict__ rmT) {
  const int idx = blockIdx.x * 256 + threadIdx.x;
  const int A = 3 * 768 * 256;
  const int B = 3 * 256 * 256;
  const int Crel = 9 * 8 * 32 * 32;
  if (idx < A) {
    int k = idx & 255;
    int n = (idx >> 8) % 768;
    int t = idx / (768 * 256);
    float v;
    if (n < 256)      v = Wq[((size_t)(t * 256 + k)) * 256 + n];
    else if (n < 512) v = Wk[((size_t)(t * 256 + k)) * 256 + (n - 256)];
    else              v = Wv[((size_t)(t * 256 + k)) * 256 + (n - 512)];
    wqkvT[idx] = f2bf(v);
  }
  int i2 = idx - A;
  if (i2 >= 0 && i2 < B) {
    int k = i2 & 255, n = (i2 >> 8) & 255, t = i2 >> 16;
    waT[i2] = f2bf(Wa[((size_t)(t * 256 + k)) * 256 + n]);
  }
  int i3 = i2 - B;
  if (i3 >= 0 && i3 < Crel) {
    int p = i3 & 31, q = (i3 >> 5) & 31, em = i3 >> 10;   // em = et*8+m
    raT[i3] = f2bf(ra[(size_t)em * 1024 + p * 32 + q]);
    rmT[i3] = f2bf(rm[(size_t)em * 1024 + p * 32 + q]);
  }
}

// ---------------- stage 1: fused QKV projection ----------------
// One block = one agent l + one 16-row tile of HW. 8 waves x 6 N-tiles = 768 cols.
// q/k/v ws layout: [m][l][hw][32] bf16 (head-major for the attention stage).
// Main loop is double-buffered: fragments for K-step kt+1 are issued before the
// WMMAs of step kt so global-load latency overlaps the matrix pipe.
__global__ __launch_bounds__(256) void qkv_kernel(
    const float* __restrict__ x, const float* __restrict__ pe,
    const unsigned short* __restrict__ wqkvT,
    const float* __restrict__ bq, const float* __restrict__ bk,
    const float* __restrict__ bv,
    unsigned short* __restrict__ qws, unsigned short* __restrict__ kws,
    unsigned short* __restrict__ vws) {
  __shared__ unsigned short Abf[16][264];   // 16 rows x 256 K, +8 pad vs LDS banks
  const int l   = blockIdx.x / NTILE_HW;
  const int hw0 = (blockIdx.x % NTILE_HW) * 16;
  const int t   = (int)pe[l * 3 + 2];

  // cooperative load + f32->bf16 convert of the activation tile
  const float* xrow = x + ((size_t)(l * HW) + hw0) * C_CH;
  for (int idx = threadIdx.x; idx < 16 * 256; idx += 256) {
    int r = idx >> 8, c = idx & 255;
    Abf[r][c] = f2bf(xrow[(size_t)r * 256 + c]);
  }
  __syncthreads();

  const int wave = threadIdx.x >> 5, lane = threadIdx.x & 31;
  const int nlo = lane & 15, hi = lane >> 4, kb = hi * 8;
  const unsigned short* wb = wqkvT + (size_t)t * 768 * 256;

  const unsigned short* wrow[6];
#pragma unroll
  for (int nn = 0; nn < 6; ++nn)
    wrow[nn] = wb + (size_t)((wave * 6 + nn) * 16 + nlo) * 256;

  v8f zero = {};
  v8f acc[6];
#pragma unroll
  for (int nn = 0; nn < 6; ++nn) acc[nn] = zero;

  // prologue: fragments for kt = 0
  v16bf a = load_frag(&Abf[nlo][0], 0, kb);
  v16bf bfr[6];
#pragma unroll
  for (int nn = 0; nn < 6; ++nn) bfr[nn] = load_frag(wrow[nn], 0, kb);

#pragma unroll
  for (int kt = 0; kt < 8; ++kt) {
    v16bf a2;
    v16bf b2[6];
    if (kt < 7) {
      a2 = load_frag(&Abf[nlo][0], (kt + 1) * 32, kb);
#pragma unroll
      for (int nn = 0; nn < 6; ++nn) b2[nn] = load_frag(wrow[nn], (kt + 1) * 32, kb);
    } else {
      a2 = a;
#pragma unroll
      for (int nn = 0; nn < 6; ++nn) b2[nn] = bfr[nn];
    }
#pragma unroll
    for (int nn = 0; nn < 6; ++nn) acc[nn] = wmma_bf16(a, bfr[nn], acc[nn]);
    a = a2;
#pragma unroll
    for (int nn = 0; nn < 6; ++nn) bfr[nn] = b2[nn];
  }

#pragma unroll
  for (int nn = 0; nn < 6; ++nn) {
    int n = (wave * 6 + nn) * 16 + nlo;
    int d = n & 255;
    float bias;
    unsigned short* dst;
    if (n < 256)      { bias = bq[t * 256 + d]; dst = qws; }
    else if (n < 512) { bias = bk[t * 256 + d]; dst = kws; }
    else              { bias = bv[t * 256 + d]; dst = vws; }
    int mh = d >> 5, p = d & 31;
    unsigned short* base = dst + ((size_t)(mh * L_AG + l) * HW) * 32 + p;
#pragma unroll
    for (int r = 0; r < 8; ++r) {         // C layout: row = r + 8*hi, col = nlo
      int row = hw0 + r + 8 * hi;
      base[(size_t)row * 32] = f2bf(acc[nn][r] + bias);
    }
  }
}

// ---------------- stage 2+3: fused bilinear attention + softmax + message combine ----
// One block = one (head m, 16-row HW tile); 1 wave. Never materializes att/v_msg in HBM.
// q/k/v tiles are staged into LDS with CDNA5 async global->LDS copies when available.
__global__ __launch_bounds__(32) void attn_kernel(
    const float* __restrict__ pe, const int* __restrict__ mask,
    const unsigned short* __restrict__ qws, const unsigned short* __restrict__ kws,
    const unsigned short* __restrict__ vws,
    const unsigned short* __restrict__ relAT, const unsigned short* __restrict__ relMT,
    unsigned short* __restrict__ ocat) {
  // 40-ushort rows: 80B stride -> 16B-aligned rows for B128 async writes AND
  // 20-dword bank stride (conflict-free 16-row fragment reads).
  __shared__ unsigned short qs[L_AG][16][40];
  __shared__ unsigned short ks[L_AG][16][40];
  __shared__ unsigned short vs[L_AG][16][40];
  __shared__ float satt[16][8];

  const int m   = blockIdx.x / NTILE_HW;
  const int hw0 = (blockIdx.x % NTILE_HW) * 16;
  const int lane = threadIdx.x;
  const int nlo = lane & 15, hi = lane >> 4, kb = hi * 8;

  int types[L_AG];
#pragma unroll
  for (int l = 0; l < L_AG; ++l) types[l] = (int)pe[l * 3 + 2];

  // stage q/k/v tiles for all 5 agents into LDS (contiguous 16x32 bf16 per agent)
#pragma unroll
  for (int l = 0; l < L_AG; ++l) {
    const unsigned short* sq = qws + ((size_t)(m * L_AG + l) * HW + hw0) * 32;
    const unsigned short* sk = kws + ((size_t)(m * L_AG + l) * HW + hw0) * 32;
    const unsigned short* sv = vws + ((size_t)(m * L_AG + l) * HW + hw0) * 32;
#if HAVE_ASYNC
#pragma unroll
    for (int half = 0; half < 2; ++half) {
      int idx = lane + half * 32;          // 64 chunks of 16B per tile
      int r = idx >> 2, c = (idx & 3) * 8; // c in ushorts, 16B aligned
      async_cp16(sq + r * 32 + c, &qs[l][r][c]);
      async_cp16(sk + r * 32 + c, &ks[l][r][c]);
      async_cp16(sv + r * 32 + c, &vs[l][r][c]);
    }
#else
    const unsigned int* uq = (const unsigned int*)sq;
    const unsigned int* uk = (const unsigned int*)sk;
    const unsigned int* uv = (const unsigned int*)sv;
    for (int idx = lane; idx < 256; idx += 32) {
      int r = idx >> 4, c2 = (idx & 15) * 2;
      *(unsigned int*)&qs[l][r][c2] = uq[idx];
      *(unsigned int*)&ks[l][r][c2] = uk[idx];
      *(unsigned int*)&vs[l][r][c2] = uv[idx];
    }
#endif
  }
#if HAVE_ASYNC
  async_wait0();
#endif
  __syncthreads();

  const float scale = 0.1767766952966369f;  // 1/sqrt(32)

  for (int i = 0; i < L_AG; ++i) {
    v16bf aq = load_frag(&qs[i][nlo][0], 0, kb);

    // ---- attention scores: (q_i @ W_att[i,j]) . k_j, per row ----
    for (int j = 0; j < L_AG; ++j) {
      int em = (types[i] * T_TYPES + types[j]) * HEADS + m;
      const unsigned short* wb = relAT + (size_t)em * 1024;
      float part[8];
#pragma unroll
      for (int r = 0; r < 8; ++r) part[r] = 0.f;
#pragma unroll
      for (int nh = 0; nh < 2; ++nh) {
        v16bf b = load_frag(wb + (nh * 16 + nlo) * 32, 0, kb);
        v8f c = {};
        c = wmma_bf16(aq, b, c);
#pragma unroll
        for (int r = 0; r < 8; ++r) {
          float kv = bf2f(ks[j][r + 8 * hi][nh * 16 + nlo]);
          part[r] += c[r] * kv;
        }
      }
      // reduce over the 16 N-lanes (stays inside each 16-lane half)
#pragma unroll
      for (int off = 1; off < 16; off <<= 1) {
#pragma unroll
        for (int r = 0; r < 8; ++r) part[r] += __shfl_xor(part[r], off, 32);
      }
      if (nlo == 0) {
#pragma unroll
        for (int r = 0; r < 8; ++r) satt[r + 8 * hi][j] = part[r] * scale;
      }
    }
    __syncthreads();

    // ---- masked softmax over j (one lane per row) ----
    if (lane < 16) {
      float s[L_AG];
      float mx = -3.0e38f;
#pragma unroll
      for (int j = 0; j < L_AG; ++j) {
        float v = satt[lane][j];
        if (mask[(size_t)(hw0 + lane) * L_AG + j] == 0) v = -3.0e38f;
        s[j] = v;
        mx = fmaxf(mx, v);
      }
      float sum = 0.f;
#pragma unroll
      for (int j = 0; j < L_AG; ++j) { s[j] = __expf(s[j] - mx); sum += s[j]; }
      float inv = 1.f / sum;
#pragma unroll
      for (int j = 0; j < L_AG; ++j) satt[lane][j] = s[j] * inv;
    }
    __syncthreads();

    // ---- out_i = sum_j att_ij * (v_j @ W_msg[i,j]) ----
    v8f acc0 = {}, acc1 = {};
    for (int j = 0; j < L_AG; ++j) {
      int em = (types[i] * T_TYPES + types[j]) * HEADS + m;
      const unsigned short* wm = relMT + (size_t)em * 1024;
      v16bf av = load_frag(&vs[j][nlo][0], 0, kb);
#pragma unroll
      for (int nh = 0; nh < 2; ++nh) {
        v16bf b = load_frag(wm + (nh * 16 + nlo) * 32, 0, kb);
        v8f c = {};
        c = wmma_bf16(av, b, c);
#pragma unroll
        for (int r = 0; r < 8; ++r) {
          float w = satt[r + 8 * hi][j];
          if (nh == 0) acc0[r] += w * c[r];
          else         acc1[r] += w * c[r];
        }
      }
    }
    // store to concatenated-head layout [l][hw][m*32+c] bf16
#pragma unroll
    for (int r = 0; r < 8; ++r) {
      size_t row = (size_t)(i * HW) + hw0 + r + 8 * hi;
      ocat[row * 256 + m * 32 + nlo]      = f2bf(acc0[r]);
      ocat[row * 256 + m * 32 + 16 + nlo] = f2bf(acc1[r]);
    }
    __syncthreads();   // satt reused next i
  }
}

// ---------------- stage 4: per-type output projection ----------------
// A tile (already bf16) staged to LDS once per block (async path when available),
// shared by all 8 waves; B fragments double-buffered across the K loop.
__global__ __launch_bounds__(256) void proj_kernel(
    const float* __restrict__ pe, const unsigned short* __restrict__ ocat,
    const unsigned short* __restrict__ waT, const float* __restrict__ ba,
    float* __restrict__ out) {
  __shared__ unsigned short At[16][264];   // 528B row stride: 16B aligned, bank-safe
  const int l   = blockIdx.x / NTILE_HW;
  const int hw0 = (blockIdx.x % NTILE_HW) * 16;
  const int t   = (int)pe[l * 3 + 2];

  const unsigned short* ag = ocat + ((size_t)(l * HW) + hw0) * 256;
#if HAVE_ASYNC
#pragma unroll
  for (int half = 0; half < 2; ++half) {
    int idx = threadIdx.x + half * 256;     // 512 chunks of 16B
    int r = idx >> 5, c = (idx & 31) * 8;   // c in ushorts, 16B aligned
    async_cp16(ag + (size_t)r * 256 + c, &At[r][c]);
  }
  async_wait0();
#else
  {
    const unsigned int* u = (const unsigned int*)ag;
    for (int idx = threadIdx.x; idx < 16 * 128; idx += 256) {
      int r = idx >> 7, c2 = (idx & 127) * 2;
      *(unsigned int*)&At[r][c2] = u[r * 128 + (idx & 127)];
    }
  }
#endif
  __syncthreads();

  const int wave = threadIdx.x >> 5, lane = threadIdx.x & 31;
  const int nlo = lane & 15, hi = lane >> 4, kb = hi * 8;

  const unsigned short* wb = waT + (size_t)t * 256 * 256;
  const unsigned short* wrow0 = wb + (size_t)((wave * 2 + 0) * 16 + nlo) * 256;
  const unsigned short* wrow1 = wb + (size_t)((wave * 2 + 1) * 16 + nlo) * 256;

  v8f acc0 = {}, acc1 = {};
  v16bf a  = load_frag(&At[nlo][0], 0, kb);
  v16bf b0 = load_frag(wrow0, 0, kb);
  v16bf b1 = load_frag(wrow1, 0, kb);
#pragma unroll
  for (int kt = 0; kt < 8; ++kt) {
    v16bf an, b0n, b1n;
    if (kt < 7) {
      an  = load_frag(&At[nlo][0], (kt + 1) * 32, kb);
      b0n = load_frag(wrow0, (kt + 1) * 32, kb);
      b1n = load_frag(wrow1, (kt + 1) * 32, kb);
    } else {
      an = a; b0n = b0; b1n = b1;
    }
    acc0 = wmma_bf16(a, b0, acc0);
    acc1 = wmma_bf16(a, b1, acc1);
    a = an; b0 = b0n; b1 = b1n;
  }
#pragma unroll
  for (int nn = 0; nn < 2; ++nn) {
    int n = (wave * 2 + nn) * 16 + nlo;
    float bias = ba[t * 256 + n];
    v8f& acc = nn ? acc1 : acc0;
#pragma unroll
    for (int r = 0; r < 8; ++r) {
      size_t row = (size_t)(l * HW) + hw0 + r + 8 * hi;
      out[row * 256 + n] = acc[r] + bias;   // out is [B,L,H,W,C]
    }
  }
}

// ---------------- workspace layout ----------------
static constexpr size_t SZ_WQKV = (size_t)3 * 768 * 256 * 2;
static constexpr size_t SZ_WA   = (size_t)3 * 256 * 256 * 2;
static constexpr size_t SZ_REL  = (size_t)9 * 8 * 32 * 32 * 2;
static constexpr size_t SZ_QKV  = (size_t)HEADS * L_AG * HW * 32 * 2;

static constexpr size_t OFF_WQKV = 0;
static constexpr size_t OFF_WA   = OFF_WQKV + SZ_WQKV;
static constexpr size_t OFF_RA   = OFF_WA + SZ_WA;
static constexpr size_t OFF_RM   = OFF_RA + SZ_REL;
static constexpr size_t OFF_Q    = OFF_RM + SZ_REL;
static constexpr size_t OFF_K    = OFF_Q + SZ_QKV;
static constexpr size_t OFF_V    = OFF_K + SZ_QKV;
static constexpr size_t OFF_OC   = OFF_V + SZ_QKV;

extern "C" void kernel_launch(void* const* d_in, const int* in_sizes, int n_in,
                              void* d_out, int out_size, void* d_ws, size_t ws_size,
                              hipStream_t stream) {
  (void)in_sizes; (void)n_in; (void)out_size; (void)ws_size;
  const float* x    = (const float*)d_in[0];
  const int*   mask = (const int*)d_in[1];
  const float* pe   = (const float*)d_in[2];
  const float* Wq   = (const float*)d_in[3];
  const float* bq   = (const float*)d_in[4];
  const float* Wk   = (const float*)d_in[5];
  const float* bk   = (const float*)d_in[6];
  const float* Wv   = (const float*)d_in[7];
  const float* bv   = (const float*)d_in[8];
  const float* Wa   = (const float*)d_in[9];
  const float* ba   = (const float*)d_in[10];
  const float* ra   = (const float*)d_in[11];
  const float* rm   = (const float*)d_in[12];
  float* out = (float*)d_out;

  char* ws = (char*)d_ws;
  unsigned short* wqkvT = (unsigned short*)(ws + OFF_WQKV);
  unsigned short* waT   = (unsigned short*)(ws + OFF_WA);
  unsigned short* raT   = (unsigned short*)(ws + OFF_RA);
  unsigned short* rmT   = (unsigned short*)(ws + OFF_RM);
  unsigned short* qws   = (unsigned short*)(ws + OFF_Q);
  unsigned short* kws   = (unsigned short*)(ws + OFF_K);
  unsigned short* vws   = (unsigned short*)(ws + OFF_V);
  unsigned short* ocat  = (unsigned short*)(ws + OFF_OC);

  const int prep_total = 3 * 768 * 256 + 3 * 256 * 256 + 9 * 8 * 32 * 32;
  prep_kernel<<<(prep_total + 255) / 256, 256, 0, stream>>>(
      Wq, Wk, Wv, Wa, ra, rm, wqkvT, waT, raT, rmT);

  qkv_kernel<<<L_AG * NTILE_HW, 256, 0, stream>>>(
      x, pe, wqkvT, bq, bk, bv, qws, kws, vws);

  attn_kernel<<<HEADS * NTILE_HW, 32, 0, stream>>>(
      pe, mask, qws, kws, vws, raT, rmT, ocat);

  proj_kernel<<<L_AG * NTILE_HW, 256, 0, stream>>>(
      pe, ocat, waT, ba, out);
}
